// crf_13022340841485
// MI455X (gfx1250) — matrix operand order repaired and checked by reading, not compiled, not run on hardware
//
#include <hip/hip_runtime.h>

typedef __attribute__((ext_vector_type(16))) _Float16 v16h;
typedef __attribute__((ext_vector_type(8)))  float    v8f;
typedef __attribute__((ext_vector_type(4)))  float    v4f;

static constexpr int   kB     = 64;
static constexpr int   kT     = 256;
static constexpr int   kK     = 128;
static constexpr float kNEG   = -10000.0f;
static constexpr float kLOG2E = 1.4426950408889634f;   // log2(e)
static constexpr float kLN2   = 0.6931471805599453f;   // ln(2)

// ---------------------------------------------------------------------------
// Kernel 1: EbtH[k][n] = (f16) exp(trans[n][k])   (transposed, exponentiated)
// In the base-2 score domain this is 2^(log2e * trans) = e^trans, unchanged.
// exp(-10000) underflows to 0 -> implements NEG transition masking exactly.
// ---------------------------------------------------------------------------
__global__ void crf_exp_trans_kernel(const float* __restrict__ trans,
                                     _Float16* __restrict__ EbtH) {
    int i = blockIdx.x * blockDim.x + threadIdx.x;   // 0 .. K*K-1
    int k = i >> 7;
    int n = i & (kK - 1);
    EbtH[i] = (_Float16)__expf(trans[n * kK + k]);
}

// ---------------------------------------------------------------------------
// Kernel 2: sequential CRF forward chain, scores kept as u = score*log2(e).
// grid = B/16 blocks of 256 threads (8 waves). Each block owns 16 batch rows.
// mask[b,t] = (t < len[b]) is monotone, so rows update UNCONDITIONALLY and
// out[b] is emitted at the freeze step t == len[b], from the row max and
// row-sum of P that stages 1/2 produce anyway. No per-step mask selects.
// ---------------------------------------------------------------------------
__launch_bounds__(256, 1)
__global__ void crf_fwd_kernel(const float* __restrict__ y,
                               const float* __restrict__ mask,
                               const _Float16* __restrict__ EbtH,
                               float* __restrict__ out) {
    constexpr int SRS = 160;  // score row stride (f32): %64==32 -> disjoint half banks
    constexpr int PRS = 136;  // P row stride (f16): 16B-aligned rows, breaks 16-way conflicts

    __shared__ float               s_score[16 * SRS];   // u = score * log2(e)
    __shared__ _Float16            s_p[16 * PRS];
    __shared__ __align__(16) float s_m[16];             // mu (base-2 row max)

    const int tid  = threadIdx.x;
    const int lane = tid & 31;
    const int wv   = tid >> 5;        // wave 0..7 -> N-tile
    const int half = lane >> 4;       // 0/1 within wave
    const int l16  = lane & 15;
    const int r    = tid >> 4;        // row 0..15 for reduction phases
    const int blk  = blockIdx.x;
    const int n0   = wv * 16;
    const int n    = n0 + l16;        // this lane's output column

    // init u: NEG*log2e everywhere except SOS column (=2)
#pragma unroll
    for (int i = 0; i < 8; ++i) {
        int k = l16 + 16 * i;
        s_score[r * SRS + k] = (k == 2) ? 0.0f : kNEG * kLOG2E;
    }

    // Row length: len[b] = sum_t mask[b,t] (mask is exact 0.0/1.0).
    float lsum = 0.0f;
#pragma unroll
    for (int i = 0; i < 16; ++i)
        lsum += mask[(blk * 16 + r) * kT + l16 * 16 + i];
#pragma unroll
    for (int off = 1; off < 16; off <<= 1)
        lsum += __shfl_xor(lsum, off, 32);
    const int lenr = (int)(lsum + 0.5f);

    // Constant B fragments (E^T) in registers.
    // B k order used here (A matches): lanes 0-15 hold K=0..15, lanes 16-31 hold K=16..31,
    // element e <-> K = 16*half + e, column N = n0 + l16.
    v16h bf[4];
#pragma unroll
    for (int c = 0; c < 4; ++c) {
#pragma unroll
        for (int e = 0; e < 16; ++e) {
            int kk = 32 * c + 16 * half + e;
            bf[c][e] = EbtH[kk * kK + n];
        }
    }
    __syncthreads();

    const long prefRow = (long)(blk * 16 + r) * kT * kK;

    for (int t = 0; t < kT; ++t) {
        // ---- stage 1: per-row max (base-2 units) via 16-lane butterfly -----
        float pm = -3.0e38f;
#pragma unroll
        for (int i = 0; i < 8; ++i)
            pm = fmaxf(pm, s_score[r * SRS + l16 + 16 * i]);
#pragma unroll
        for (int off = 1; off < 16; off <<= 1)
            pm = fmaxf(pm, __shfl_xor(pm, off, 32));
        if (l16 == 0)
            s_m[r] = pm;

        // ---- stage 2: P = 2^(u - mu) -> LDS f16; row sum for lse -----------
        float rsum = 0.0f;
#pragma unroll
        for (int i = 0; i < 8; ++i) {
            int k = l16 + 16 * i;
            float p = __builtin_amdgcn_exp2f(s_score[r * SRS + k] - pm);
            s_p[r * PRS + k] = (_Float16)p;
            rsum += p;
        }
#pragma unroll
        for (int off = 1; off < 16; off <<= 1)
            rsum += __shfl_xor(rsum, off, 32);
        // Freeze point: u at loop-top of iteration t is the score after t
        // masked updates, so t == len[b] is exactly the reference's output.
        if (l16 == 0 && t == lenr)
            out[blk * 16 + r] = kLN2 * (pm + __builtin_amdgcn_logf(rsum));

        if (t + 1 < kT)
            __builtin_prefetch(&y[prefRow + (long)(t + 1) * kK + l16 * 8], 0, 3);
        __syncthreads();  // barrier A: P visible to all waves

        // ---- stage 3: load all A fragments, then WMMA chain ----------------
        // A k order used here (matches B): half 0 -> K 0-15, half 1 -> K 16-31
        v16h af[4];
#pragma unroll
        for (int c = 0; c < 4; ++c) {
#pragma unroll
            for (int e = 0; e < 16; ++e) {
                int kk = 16 * half + e;
                af[c][e] = s_p[l16 * PRS + 32 * c + kk];
            }
        }
        v8f acc0 = {};
        v8f acc1 = {};
        acc0 = __builtin_amdgcn_wmma_f32_16x16x32_f16(false, af[0], false, bf[0],
                                                      (short)0, acc0, false, false);
        acc1 = __builtin_amdgcn_wmma_f32_16x16x32_f16(false, af[1], false, bf[1],
                                                      (short)0, acc1, false, false);
        acc0 = __builtin_amdgcn_wmma_f32_16x16x32_f16(false, af[2], false, bf[2],
                                                      (short)0, acc0, false, false);
        acc1 = __builtin_amdgcn_wmma_f32_16x16x32_f16(false, af[3], false, bf[3],
                                                      (short)0, acc1, false, false);
        v8f acc = acc0 + acc1;

        // ---- stage 4: u' = y*log2e + mu + log2(C); unconditional store -----
        // C/D layout: lane half h, VGPR v -> row M = v + 8h, col N = n0 + l16
        const v4f* mp = (const v4f*)(&s_m[8 * half]);
        v4f mlo = mp[0];
        v4f mhi = mp[1];
        float mu8[8] = {mlo.x, mlo.y, mlo.z, mlo.w, mhi.x, mhi.y, mhi.z, mhi.w};
#pragma unroll
        for (int v = 0; v < 8; ++v) {
            int bb = blk * 16 + (v + 8 * half);
            float s = y[((long)bb * kT + t) * kK + n] * kLOG2E + mu8[v] +
                      __builtin_amdgcn_logf(acc[v]);      // raw v_log_f32 (log2)
            s_score[(v + 8 * half) * SRS + n] = s;
        }
        __syncthreads();  // barrier B: score update visible before next step
    }

    // ---- epilogue: rows with len == T emit from u^(T) ----------------------
    float pm = -3.0e38f;
#pragma unroll
    for (int i = 0; i < 8; ++i)
        pm = fmaxf(pm, s_score[r * SRS + l16 + 16 * i]);
#pragma unroll
    for (int off = 1; off < 16; off <<= 1)
        pm = fmaxf(pm, __shfl_xor(pm, off, 32));
    float sum = 0.0f;
#pragma unroll
    for (int i = 0; i < 8; ++i)
        sum += __builtin_amdgcn_exp2f(s_score[r * SRS + l16 + 16 * i] - pm);
#pragma unroll
    for (int off = 1; off < 16; off <<= 1)
        sum += __shfl_xor(sum, off, 32);
    if (l16 == 0 && lenr == kT)
        out[blk * 16 + r] = kLN2 * (pm + __builtin_amdgcn_logf(sum));
}

// ---------------------------------------------------------------------------
extern "C" void kernel_launch(void* const* d_in, const int* in_sizes, int n_in,
                              void* d_out, int out_size, void* d_ws, size_t ws_size,
                              hipStream_t stream) {
    const float* y     = (const float*)d_in[0];   // (B,T,K) f32
    const float* maskp = (const float*)d_in[1];   // (B,T)   f32
    const float* trans = (const float*)d_in[2];   // (K,K)   f32
    float* out = (float*)d_out;                   // (B,)    f32

    _Float16* EbtH = (_Float16*)d_ws;             // K*K f16 = 32 KB scratch

    crf_exp_trans_kernel<<<(kK * kK) / 256, 256, 0, stream>>>(trans, EbtH);
    crf_fwd_kernel<<<kB / 16, 256, 0, stream>>>(y, maskp, EbtH, out);
}
